// BaseMultivariateHawkes_36782099923573
// MI455X (gfx1250) — matrix operand order repaired and checked by reading, not compiled, not run on hardware
//
#include <hip/hip_runtime.h>
#include <hip/hip_bf16.h>

typedef __attribute__((ext_vector_type(16))) _Float16 v16h;
typedef __attribute__((ext_vector_type(8)))  _Float16 v8h;
typedef __attribute__((ext_vector_type(8)))  float    v8f;

#define N_EVENTS_C 4096
#define N_SPACE_C  64
#define INT_RES_C  100
#define T1_C       100.0f
#define HSTRIDE    40          // s_H row stride in halfs: 80B, 16B-aligned, conflict-free
#define ASTRIDE    72          // s_alpha f16 row stride in halfs: 144B, 16B-aligned

// f16 A-matrix 16x32 element -> K mapping: lanes<16 hold K {0..7,16..23},
// lanes>=16 hold K {8..15,24..31}.
__device__ __forceinline__ int kmap16(int e, int hi) {
    return (e < 8) ? (hi * 8 + e) : (8 + hi * 8 + e);
}
__device__ __forceinline__ float rfl(float x) {   // scalarize a wave-uniform value
    return __int_as_float(__builtin_amdgcn_readfirstlane(__float_as_int(x)));
}
__device__ __forceinline__ v16h cat16(v8h lo, v8h hic) {
    return __builtin_shufflevector(lo, hic, 0,1,2,3,4,5,6,7,8,9,10,11,12,13,14,15);
}
#define WMMA_F16(A, B, C) __builtin_amdgcn_wmma_f32_16x16x32_f16(false, (A), false, (B), (short)0, (C), false, false)

// gfx1250 async global->LDS copy: 16B per lane (512B per wave-instruction),
// GVS mode: mem = SGPR base + 32-bit VGPR offset; LDS dest addr from VGPR.
__device__ __forceinline__ void async_cp_512(unsigned lds_addr, unsigned gofs,
                                             const void* base) {
    asm volatile("global_load_async_to_lds_b128 %0, %1, %2"
                 :: "v"(lds_addr), "v"(gofs), "s"(base) : "memory");
}
__device__ __forceinline__ void async_wait0() {
    asm volatile("s_wait_asynccnt 0x0" ::: "memory");
}

// ---------------------------------------------------------------------------
// term1: 128 blocks x 2 waves. 32 event rows per block (two 16-row A tiles);
// K tiles split odd/even across the waves. G = E*H: A built rank-1 (row-factor
// x f16 column-factor, v_pk_mul), H is a one-hot f16 LDS image updated with
// 2 ds_store_b16 per tile; all 4 B tiles prefetched (8 ds_load_b128, single
// wait) then 8 WMMAs issue back-to-back. val_i = alpha[sp_i,:].G[i,:].
// ---------------------------------------------------------------------------
__global__ __launch_bounds__(64)
void hawkes_term1_kernel(const float* __restrict__ data,
                         const float* __restrict__ mu,
                         const float* __restrict__ alpha,
                         const float* __restrict__ beta_p,
                         float* __restrict__ partial) {
    __shared__ __align__(16) float s_d[2 * N_EVENTS_C];   // interleaved (t, sp)
    __shared__ __align__(16) _Float16 s_H[2][N_SPACE_C][HSTRIDE];
    __shared__ __align__(16) _Float16 s_cf[2][32];
    __shared__ float s_G[32][N_SPACE_C + 1];
    __shared__ float s_red[32];

    const int tid  = threadIdx.x;
    const int wave = tid >> 5;
    const int lane = tid & 31;
    const int hi   = lane >> 4;
    const int lr   = lane & 15;
    const int rowBase = blockIdx.x * 32;
    const float beta = beta_p[0];
    __builtin_prefetch(alpha, 0, 3);              // global_prefetch_b8

    const int nK    = rowBase / 32 + 1;           // triangular: j < rowBase+32
    const int nLoad = nK * 32;

    // async stage the (t, sp) pairs, 64 events (512B) per instruction
    {
        const int nInst = (nLoad + 63) / 64;
        const unsigned lbase = (unsigned)(uintptr_t)&s_d[0] + (unsigned)(lane * 16);
        for (int q = wave; q < nInst; q += 2)
            async_cp_512(lbase + (unsigned)(q * 512), (unsigned)(q * 512 + lane * 16), data);
        async_wait0();
    }
    for (int k = tid; k < 2 * N_SPACE_C * HSTRIDE / 2; k += 64) ((int*)s_H)[k] = 0;
    __syncthreads();

    const int   i0 = rowBase + lr, i1 = rowBase + 16 + lr;
    const float t0 = s_d[2 * i0], t1 = s_d[2 * i1];

    v8f acc[2][4] = {};                           // partial G: 2 row tiles x 4 col tiles
    int sp_old = 0;

    for (int kb = wave; kb < nK; kb += 2) {       // per-wave K stripe
        const int   jbase = kb * 32;
        const float tref  = rfl(s_d[2 * (jbase + 31)]);   // tile max (sorted)
        const float tmin  = rfl(s_d[2 * jbase]);

        const float2 ev = ((const float2*)s_d)[jbase + lane];   // one ds_load_b64
        const int sp_new = (int)ev.y;
        s_cf[wave][lane]        = (_Float16)__expf(-beta * (tref - ev.x));
        s_H[wave][sp_old][lane] = (_Float16)0.0f;
        s_H[wave][sp_new][lane] = (_Float16)1.0f;
        sp_old = sp_new;
        __builtin_amdgcn_wave_barrier();          // ordering only; LDS in-order per wave

        const _Float16 rf0 = (_Float16)(beta * __expf(-beta * (t0 - tref)));
        const _Float16 rf1 = (_Float16)(beta * __expf(-beta * (t1 - tref)));

        const v8h* cfp = (const v8h*)s_cf[wave];
        const v16h cfv = cat16(cfp[hi], cfp[2 + hi]);   // cf[kmap16(e,hi)]

        v16h a0, a1;
        if ((kb < nK - 1) && (tmin > 0.0f)) {     // interior: mask all-true, pure pk_mul
            a0 = cfv * rf0;
            a1 = cfv * rf1;
        } else {                                  // straddling tile: per-element mask
#pragma unroll
            for (int e = 0; e < 16; ++e) {
                const int   j  = jbase + kmap16(e, hi);
                const float tj = s_d[2 * j];
                const float cf = (float)cfv[e];
                const bool  ok = (tj != 0.0f);
                a0[e] = (_Float16)(((j < i0) && ok) ? (float)rf0 * cf : 0.0f);
                a1[e] = (_Float16)(((j < i1) && ok) ? (float)rf1 * cf : 0.0f);
            }
        }

        v16h b[4];                                // prefetch all 4 B tiles, single wait
#pragma unroll
        for (int c = 0; c < 4; ++c) {
            const v8h* hp = (const v8h*)&s_H[wave][c * 16 + lr][0];
            b[c] = cat16(hp[hi], hp[2 + hi]);
        }
#pragma unroll
        for (int c = 0; c < 4; ++c) {             // 8 WMMAs back-to-back
            acc[0][c] = WMMA_F16(a0, b[c], acc[0][c]);
            acc[1][c] = WMMA_F16(a1, b[c], acc[1][c]);
        }
    }

    // merge the two waves' partial G through LDS
    if (wave == 0) {
#pragma unroll
        for (int r = 0; r < 2; ++r)
#pragma unroll
            for (int c = 0; c < 4; ++c)
#pragma unroll
                for (int v = 0; v < 8; ++v)
                    s_G[r * 16 + (hi ? v + 8 : v)][c * 16 + lr] = acc[r][c][v];
    }
    __syncthreads();
    if (wave == 1) {
#pragma unroll
        for (int r = 0; r < 2; ++r)
#pragma unroll
            for (int c = 0; c < 4; ++c)
#pragma unroll
                for (int v = 0; v < 8; ++v)
                    s_G[r * 16 + (hi ? v + 8 : v)][c * 16 + lr] += acc[r][c][v];
    }
    __syncthreads();

    if (tid < 32) {                               // one event row per thread
        const int   ii  = rowBase + tid;
        const int   spi = (int)s_d[2 * ii + 1];
        const float* arow = alpha + spi * N_SPACE_C;
        float val = 0.0f;
#pragma unroll 8
        for (int s = 0; s < N_SPACE_C; ++s) val += arow[s] * s_G[tid][s];
        const float lam = fmaxf(mu[spi] + val, 0.0f);
        s_red[tid] = __logf(lam);
    }
    __syncthreads();
    if (tid == 0) {
        float sum = 0.0f;
#pragma unroll
        for (int t = 0; t < 32; ++t) sum += s_red[t];
        partial[blockIdx.x] = sum;
    }
}

// ---------------------------------------------------------------------------
// term2: 7 blocks x 1 wave x 16 grid rows. Stage 1: F = e*H (same LDS one-hot
// mechanics, async staging, early uniform break). Stage 2: val = F x alpha^T
// with alpha pre-converted to f16 in LDS (B tiles = ds_load_b128).
// ---------------------------------------------------------------------------
__global__ __launch_bounds__(32)
void hawkes_term2_kernel(const float* __restrict__ data,
                         const float* __restrict__ mu,
                         const float* __restrict__ alpha,
                         const float* __restrict__ beta_p,
                         float* __restrict__ partial) {
    __shared__ __align__(16) float s_d[2 * N_EVENTS_C];
    __shared__ __align__(16) _Float16 s_H[N_SPACE_C][HSTRIDE];
    __shared__ __align__(16) _Float16 s_cf[32];
    __shared__ __align__(16) _Float16 s_ah[N_SPACE_C][ASTRIDE];
    __shared__ float s_F[16][N_SPACE_C + 1];
    __shared__ float s_red[32];

    const int lane = threadIdx.x;
    const int hi   = lane >> 4;
    const int lr   = lane & 15;
    const int rowBase = blockIdx.x * 16;
    const float beta = beta_p[0];
    const float step = T1_C / (float)(INT_RES_C - 1);

    {
        const unsigned lbase = (unsigned)(uintptr_t)&s_d[0] + (unsigned)(lane * 16);
        for (int q = 0; q < 2 * N_EVENTS_C * 4 / 512; ++q)
            async_cp_512(lbase + (unsigned)(q * 512), (unsigned)(q * 512 + lane * 16), data);
        async_wait0();
    }
    for (int x = lane; x < N_SPACE_C * N_SPACE_C; x += 32)
        s_ah[x >> 6][x & 63] = (_Float16)alpha[x];
    for (int k = lane; k < N_SPACE_C * HSTRIDE / 2; k += 32) ((int*)s_H)[k] = 0;
    __syncthreads();

    const int   g  = rowBase + lr;                // rows >= 100 discarded later
    const float tg = (float)g * step;
    const int   gmaxi = (rowBase + 15 < INT_RES_C - 1) ? rowBase + 15 : INT_RES_C - 1;
    const float tgmax = (float)gmaxi * step;
    const float tgmin = (float)rowBase * step;

    v8f acc[4] = {};
    int sp_old = 0;

    for (int kb = 0; kb < N_EVENTS_C / 32; ++kb) {
        const int   jbase = kb * 32;
        const float tmin  = rfl(s_d[2 * jbase]);
        if (tmin >= tgmax) break;                 // sorted: nothing left contributes
        const float tref  = rfl(s_d[2 * (jbase + 31)]);

        const float2 ev = ((const float2*)s_d)[jbase + lane];
        const int sp_new = (int)ev.y;
        s_cf[lane]        = (_Float16)__expf(-beta * (tref - ev.x));
        s_H[sp_old][lane] = (_Float16)0.0f;
        s_H[sp_new][lane] = (_Float16)1.0f;
        sp_old = sp_new;
        __builtin_amdgcn_wave_barrier();

        const _Float16 rf = (_Float16)(beta * __expf(-beta * (tg - tref)));
        const v8h* cfp = (const v8h*)s_cf;
        const v16h cfv = cat16(cfp[hi], cfp[2 + hi]);

        v16h a;
        if ((tref < tgmin) && (tmin > 0.0f)) {    // tile fully before all grid rows
            a = cfv * rf;
        } else {
#pragma unroll
            for (int e = 0; e < 16; ++e) {
                const float tj = s_d[2 * (jbase + kmap16(e, hi))];
                a[e] = (_Float16)(((tj < tg) && (tj != 0.0f)) ? (float)rf * (float)cfv[e] : 0.0f);
            }
        }

        v16h b[4];
#pragma unroll
        for (int c = 0; c < 4; ++c) {
            const v8h* hp = (const v8h*)&s_H[c * 16 + lr][0];
            b[c] = cat16(hp[hi], hp[2 + hi]);
        }
#pragma unroll
        for (int c = 0; c < 4; ++c)
            acc[c] = WMMA_F16(a, b[c], acc[c]);
    }

#pragma unroll
    for (int c = 0; c < 4; ++c)
#pragma unroll
        for (int v = 0; v < 8; ++v)
            s_F[hi ? v + 8 : v][c * 16 + lr] = acc[c][v];
    __syncthreads();

    // Stage 2: val(16x64) = F(16x64) x alpha^T(64x64), 2 K-tiles x 4 col-tiles
    v8f vacc[4] = {};
#pragma unroll
    for (int kt = 0; kt < 2; ++kt) {
        v16h a;
#pragma unroll
        for (int e = 0; e < 16; ++e)
            a[e] = (_Float16)s_F[lr][kt * 32 + kmap16(e, hi)];
#pragma unroll
        for (int c = 0; c < 4; ++c) {
            const v8h* ap = (const v8h*)&s_ah[c * 16 + lr][0];  // row s of alpha (f16)
            const v16h b  = cat16(ap[kt * 4 + hi], ap[kt * 4 + 2 + hi]);
            vacc[c] = WMMA_F16(a, b, vacc[c]);
        }
    }

    float myv = 0.0f;
#pragma unroll
    for (int c = 0; c < 4; ++c) {
        const int   s  = c * 16 + lr;
        const float ms = mu[s];
#pragma unroll
        for (int v = 0; v < 8; ++v) {
            const int gg = rowBase + (hi ? v + 8 : v);
            const float l = fmaxf(ms + vacc[c][v], 0.0f);
            myv += (gg < INT_RES_C) ? l : 0.0f;
        }
    }
    s_red[lane] = myv;
    __syncthreads();
    if (lane == 0) {
        float sum = 0.0f;
#pragma unroll
        for (int t = 0; t < 32; ++t) sum += s_red[t];
        partial[blockIdx.x] = sum;
    }
}

// Deterministic final reduction: loglik = term1 - term2 * T1/INT_RES
__global__ void hawkes_reduce_kernel(const float* __restrict__ partial,
                                     float* __restrict__ out) {
    if (threadIdx.x == 0 && blockIdx.x == 0) {
        float s1 = 0.0f;
        for (int b = 0; b < 128; ++b) s1 += partial[b];
        float s2 = 0.0f;
        for (int b = 0; b < 7; ++b) s2 += partial[128 + b];
        out[0] = s1 - s2 * (T1_C / (float)INT_RES_C);
    }
}

extern "C" void kernel_launch(void* const* d_in, const int* in_sizes, int n_in,
                              void* d_out, int out_size, void* d_ws, size_t ws_size,
                              hipStream_t stream) {
    const float* data   = (const float*)d_in[0];   // (4096, 2) [t, space]
    const float* mu     = (const float*)d_in[1];   // (64,)
    const float* alpha  = (const float*)d_in[2];   // (64, 64)
    const float* beta_p = (const float*)d_in[3];   // scalar
    float* ws  = (float*)d_ws;                     // 135 floats of partials
    float* out = (float*)d_out;

    hipLaunchKernelGGL(hawkes_term1_kernel, dim3(N_EVENTS_C / 32), dim3(64), 0, stream,
                       data, mu, alpha, beta_p, ws);
    hipLaunchKernelGGL(hawkes_term2_kernel, dim3((INT_RES_C + 15) / 16), dim3(32), 0, stream,
                       data, mu, alpha, beta_p, ws + 128);
    hipLaunchKernelGGL(hawkes_reduce_kernel, dim3(1), dim3(32), 0, stream, ws, out);
}
